// MoETop2TwoExpertsPerRank_58746562675158
// MI455X (gfx1250) — compile-verified, matrix operand order checked by reading, not run on hardware
//
#include <hip/hip_runtime.h>

// MoE Top-2, two experts per rank, world_size=1:
//   y = c0(t)*FFN0(x_t) + c1(t)*FFN1(x_t),  c_e(t) = sum_s w[t,s]*[id[t,s]%2==e]
// Compute-bound (1.1 TFLOP, AI~1700) -> bf16 WMMA 16x16x32, f32 accumulate.
// All operands pre-converted (and weights pre-transposed) to bf16 in workspace,
// so the GEMM hot loop is pure b128 copies + ds_load_b128 + v_wmma.

#define DM   2048
#define DFF  8192
#define NTOK 8192

typedef __attribute__((ext_vector_type(16))) __bf16       v16bf;
typedef __attribute__((ext_vector_type(8)))  float        v8f;
typedef __attribute__((ext_vector_type(4)))  float        f32x4;
typedef __attribute__((ext_vector_type(4)))  unsigned int u32x4;
typedef __attribute__((ext_vector_type(2)))  unsigned int u32x2;

union FragU { u32x4 q[2]; v16bf v; };

__device__ __forceinline__ unsigned short f2bf(float f) {
  unsigned u = __float_as_uint(f);
  u = u + 0x7FFFu + ((u >> 16) & 1u);  // round-to-nearest-even
  return (unsigned short)(u >> 16);
}
__device__ __forceinline__ unsigned f2bf_pk(float a, float b) {
  return (unsigned)f2bf(a) | ((unsigned)f2bf(b) << 16);
}

// ---- per-token gate coefficients: coef[e*NTOK + t] ----
__global__ void coef_kernel(const int* __restrict__ ids, const float* __restrict__ w,
                            float* __restrict__ coef) {
  int t = blockIdx.x * blockDim.x + threadIdx.x;
  if (t < NTOK) {
    float c0 = 0.f, c1 = 0.f;
#pragma unroll
    for (int s = 0; s < 2; ++s) {
      int   e  = ids[t * 2 + s] & 1;
      float ww = w[t * 2 + s];
      c0 += (e == 0) ? ww : 0.f;
      c1 += (e == 1) ? ww : 0.f;
    }
    coef[t]        = c0;
    coef[NTOK + t] = c1;
  }
}

// ---- fp32 -> bf16, contiguous (for x) ----
__global__ void f32_to_bf16(const float* __restrict__ src, unsigned short* __restrict__ dst) {
  int i = (blockIdx.x * blockDim.x + threadIdx.x) * 4;
  f32x4 v = *(const f32x4*)(src + i);
  u32x2 p;
  p.x = f2bf_pk(v.x, v.y);
  p.y = f2bf_pk(v.z, v.w);
  *(u32x2*)(dst + i) = p;
}

// ---- W[K][N] fp32 -> WT[N][K] bf16, 64x64 LDS tile ----
__global__ __launch_bounds__(256) void transpose_f32_to_bf16(
    const float* __restrict__ W, unsigned short* __restrict__ WT, int K, int N) {
  __shared__ unsigned short t[64 * 64];
  const int k0 = blockIdx.y * 64, n0 = blockIdx.x * 64;
  const int tid = threadIdx.x;
#pragma unroll
  for (int i = 0; i < 4; ++i) {
    int f = i * 256 + tid;
    int kr = f >> 4, c4 = (f & 15) << 2;  // 16 float4 per k-row
    f32x4 v = *(const f32x4*)(W + (size_t)(k0 + kr) * N + n0 + c4);
    t[(c4 + 0) * 64 + kr] = f2bf(v.x);
    t[(c4 + 1) * 64 + kr] = f2bf(v.y);
    t[(c4 + 2) * 64 + kr] = f2bf(v.z);
    t[(c4 + 3) * 64 + kr] = f2bf(v.w);
  }
  __syncthreads();
#pragma unroll
  for (int i = 0; i < 2; ++i) {
    int f = i * 256 + tid;
    int nn = f >> 3, c8 = (f & 7) << 3;  // 8 chunks of 8 bf16 per out-row
    *(u32x4*)(WT + (size_t)(n0 + nn) * K + k0 + c8) = *(const u32x4*)&t[nn * 64 + c8];
  }
}

// ---- tiled GEMM: block 256x128, BK=32, 8 wave32s (4x2), 64x64 per wave ----
// A bf16 [M][K], BT bf16 [N][K] (pre-transposed). 16 WMMA per wave per K-step.
// MODE 0: epilogue = bias + exact GELU -> bf16 H
// MODE 1: epilogue = coef*(acc+bias) -> Out  =
// MODE 2: epilogue = coef*(acc+bias) -> Out +=
template <int MODE>
__global__ __launch_bounds__(256) void moe_gemm(
    const unsigned short* __restrict__ A, const unsigned short* __restrict__ BT,
    const float* __restrict__ bias, unsigned short* __restrict__ Hout,
    const float* __restrict__ coefE, float* __restrict__ Out,
    int ldk, int ldc, int nk) {
  __shared__ unsigned short lds_a[2][256 * 32];  // [row][k]
  __shared__ unsigned short lds_b[2][128 * 32];  // [col][k]

  const int tid  = threadIdx.x;
  const int lane = tid & 31;
  const int wid  = tid >> 5;
  const int wm   = (wid & 3) << 6;  // 0/64/128/192
  const int wn   = (wid >> 2) << 6; // 0/64
  const int tm   = blockIdx.y << 8; // *256
  const int tn   = blockIdx.x << 7; // *128

  v8f acc[4][4];
#pragma unroll
  for (int i = 0; i < 4; ++i)
#pragma unroll
    for (int j = 0; j < 4; ++j)
#pragma unroll
      for (int r = 0; r < 8; ++r) acc[i][j][r] = 0.f;

  u32x4 rga[4], rgb[2];

  auto fetch = [&](int kt) {
    const int kb = kt << 5;
#pragma unroll
    for (int i = 0; i < 4; ++i) {  // A tile 256x32: 4 chunks/row of 8 bf16
      int f = i * 256 + tid;
      int r = f >> 2, c8 = (f & 3) << 3;
      rga[i] = *(const u32x4*)(A + (size_t)(tm + r) * ldk + kb + c8);
    }
#pragma unroll
    for (int i = 0; i < 2; ++i) {  // B tile 128x32
      int f = i * 256 + tid;
      int r = f >> 2, c8 = (f & 3) << 3;
      rgb[i] = *(const u32x4*)(BT + (size_t)(tn + r) * ldk + kb + c8);
    }
    if (kt + 2 < nk) {  // L2 prefetch of K+2 tiles
      __builtin_prefetch(BT + (size_t)(tn + (tid >> 1)) * ldk + ((kt + 2) << 5), 0, 1);
      __builtin_prefetch(A + (size_t)(tm + tid) * ldk + ((kt + 2) << 5), 0, 1);
    }
  };

  auto stage = [&](int d) {
#pragma unroll
    for (int i = 0; i < 4; ++i) {
      int f = i * 256 + tid;
      int r = f >> 2, c8 = (f & 3) << 3;
      *(u32x4*)&lds_a[d][r * 32 + c8] = rga[i];
    }
#pragma unroll
    for (int i = 0; i < 2; ++i) {
      int f = i * 256 + tid;
      int r = f >> 2, c8 = (f & 3) << 3;
      *(u32x4*)&lds_b[d][r * 32 + c8] = rgb[i];
    }
  };

  auto compute = [&](int s) {
    const int row = lane & 15;
    const int k0  = (lane >> 4) << 3;  // 16-bit A/B layout: K-base = (lane/16)*8
    FragU a[4];
#pragma unroll
    for (int i = 0; i < 4; ++i) {  // two ds_load_b128 per fragment
      const unsigned short* p = &lds_a[s][(wm + i * 16 + row) * 32 + k0];
      a[i].q[0] = *(const u32x4*)p;
      a[i].q[1] = *(const u32x4*)(p + 16);  // K+16..K+23 half
    }
#pragma unroll
    for (int j = 0; j < 4; ++j) {
      FragU b;
      const unsigned short* p = &lds_b[s][(wn + j * 16 + row) * 32 + k0];
      b.q[0] = *(const u32x4*)p;
      b.q[1] = *(const u32x4*)(p + 16);
#pragma unroll
      for (int i = 0; i < 4; ++i)
        acc[i][j] = __builtin_amdgcn_wmma_f32_16x16x32_bf16(
            false, a[i].v, false, b.v, (short)0, acc[i][j], false, false);
    }
  };

  // software-pipelined main loop (double-buffered LDS)
  fetch(0);
  stage(0);
  int s = 0;
  for (int kt = 0; kt < nk; ++kt) {
    __syncthreads();
    if (kt + 1 < nk) fetch(kt + 1);
    compute(s);
    if (kt + 1 < nk) stage(s ^ 1);
    s ^= 1;
  }

  // epilogue; D layout: VGPR r -> M = r + 8*(lane/16), N = lane%16
#pragma unroll
  for (int j = 0; j < 4; ++j) {
    int   n  = tn + wn + j * 16 + (lane & 15);
    float bv = bias[n];
#pragma unroll
    for (int i = 0; i < 4; ++i) {
      int mbase = tm + wm + i * 16 + ((lane >> 4) << 3);
#pragma unroll
      for (int r = 0; r < 8; ++r) {
        float v = acc[i][j][r] + bv;
        if (MODE == 0) {
          v = 0.5f * v * (1.0f + erff(v * 0.70710678118654752f));  // exact GELU
          Hout[(size_t)(mbase + r) * DFF + n] = f2bf(v);
        } else {
          float  res = coefE[mbase + r] * v;
          size_t o   = (size_t)(mbase + r) * ldc + n;
          if (MODE == 2) res += Out[o];
          Out[o] = res;
        }
      }
    }
  }
}

extern "C" void kernel_launch(void* const* d_in, const int* in_sizes, int n_in,
                              void* d_out, int out_size, void* d_ws, size_t ws_size,
                              hipStream_t stream) {
  const float* x   = (const float*)d_in[0];
  const int*   ids = (const int*)d_in[1];
  const float* gw  = (const float*)d_in[2];
  const float* W1[2] = {(const float*)d_in[3], (const float*)d_in[7]};
  const float* b1[2] = {(const float*)d_in[4], (const float*)d_in[8]};
  const float* W2[2] = {(const float*)d_in[5], (const float*)d_in[9]};
  const float* b2[2] = {(const float*)d_in[6], (const float*)d_in[10]};
  float* out = (float*)d_out;

  // workspace: coef | xb (bf16 x) | W1T | W2T (per-expert, reused) | H
  char*  p    = (char*)d_ws;
  float* coef = (float*)p;                 p += (((size_t)2 * NTOK * 4) + 255) & ~(size_t)255;
  unsigned short* xb  = (unsigned short*)p; p += (size_t)NTOK * DM * 2;
  unsigned short* W1T = (unsigned short*)p; p += (size_t)DM * DFF * 2;
  unsigned short* W2T = (unsigned short*)p; p += (size_t)DFF * DM * 2;
  unsigned short* H   = (unsigned short*)p;

  dim3 blk(256, 1, 1);
  coef_kernel<<<NTOK / 256, blk, 0, stream>>>(ids, gw, coef);
  f32_to_bf16<<<(NTOK * DM) / 1024, blk, 0, stream>>>(x, xb);

  dim3 gT1(DFF / 64, DM / 64, 1);      // W1: K=DM,  N=DFF
  dim3 gT2(DM / 64, DFF / 64, 1);      // W2: K=DFF, N=DM
  dim3 g1(DFF / 128, NTOK / 256, 1);   // gemm1 tiles
  dim3 g2(DM / 128, NTOK / 256, 1);    // gemm2 tiles

  // expert 0
  transpose_f32_to_bf16<<<gT1, blk, 0, stream>>>(W1[0], W1T, DM, DFF);
  transpose_f32_to_bf16<<<gT2, blk, 0, stream>>>(W2[0], W2T, DFF, DM);
  moe_gemm<0><<<g1, blk, 0, stream>>>(xb, W1T, b1[0], H, nullptr, nullptr, DM, 0, DM / 32);
  moe_gemm<1><<<g2, blk, 0, stream>>>(H, W2T, b2[0], nullptr, coef, out, DFF, DM, DFF / 32);
  // expert 1
  transpose_f32_to_bf16<<<gT1, blk, 0, stream>>>(W1[1], W1T, DM, DFF);
  transpose_f32_to_bf16<<<gT2, blk, 0, stream>>>(W2[1], W2T, DFF, DM);
  moe_gemm<0><<<g1, blk, 0, stream>>>(xb, W1T, b1[1], H, nullptr, nullptr, DM, 0, DM / 32);
  moe_gemm<2><<<g2, blk, 0, stream>>>(H, W2T, b2[1], nullptr, coef + NTOK, out, DFF, DM, DFF / 32);
}
